// IFNet_time_61340722922005
// MI455X (gfx1250) — matrix operand order repaired and checked by reading, not compiled, not run on hardware
//
#include <hip/hip_runtime.h>
#include <hip/hip_bf16.h>
#include <stdint.h>

typedef __bf16 bf16;
typedef bf16  v16bf __attribute__((ext_vector_type(16)));
typedef float v8f   __attribute__((ext_vector_type(8)));
typedef unsigned int u32x4 __attribute__((ext_vector_type(4)));
typedef int i32x4 __attribute__((ext_vector_type(4)));
typedef int i32x8 __attribute__((ext_vector_type(8)));

constexpr int HH = 224, WW = 224;
constexpr int HWp = HH * WW;                 // 50176 pixels
constexpr int NF  = 32;
constexpr int INF = 100;                     // conv channels
constexpr int PAD = 226;                     // padded H/W (1 guard each side)
constexpr int CHS = PAD * PAD;               // 51076: padded channel slice
constexpr int KCONV    = INF * 9;            // 900
constexpr int KSTEPS   = (KCONV + 31) / 32;  // 29
constexpr int OCT_P    = 8;                  // oc padded 100 -> 128 (4 tiles x 2 groups)
constexpr int PIXT     = HWp / 16;           // 3136
constexpr int KLAST    = 5 * INF;            // 500
constexpr int KSTEPS_L = (KLAST + 31) / 32;  // 16
constexpr int OC36     = NF + 4;             // 36
constexpr int OCT_L    = 3;                  // pad 36 -> 48
constexpr int H2 = 2 * HH, W2 = 2 * WW;
constexpr int S2 = H2 * W2;                  // 200704

constexpr int LDS_TILE = INF * 3 * PAD;      // 67800 bf16: [ic][ky][226]
constexpr int ZOFF     = LDS_TILE;           // zero slot for K>=900 guard
constexpr int LDS_ELEMS = LDS_TILE + 256;
constexpr int LDS_BYTES = LDS_ELEMS * 2;     // 136112 B (< 320KB WGP LDS)

__device__ __forceinline__ int iclamp(int v, int lo, int hi) {
  return v < lo ? lo : (v > hi ? hi : v);
}

// K index held by element e of the 16-bit A fragment in a given lane
// (ISA 05_wmma 7.12.2: 16-bit A 16x32 layout).
__device__ __forceinline__ int a_elem_k(int lane, int e) {
  int kgroup = lane >> 4;
  int v = e >> 1;
  int base = (v < 4 ? 0 : 16) + ((v & 3) << 1);
  return base + (kgroup << 3) + (e & 1);
}

// LDS offset for conv-K index k: (ic*3+ky)*226 + kx == (k/3)*223 + k
__device__ __forceinline__ int conv_off(int k) {
  unsigned q = (unsigned)k / 3u;
  return (int)(q * 223u + (unsigned)k);
}

// ---------------- weight packing into A-fragment layout ----------------

__global__ void pack_conv_w(const float* __restrict__ w_head,
                            const float* __restrict__ w_blocks,
                            bf16* __restrict__ apack) {
  int t = blockIdx.x * blockDim.x + threadIdx.x;
  int total = 5 * OCT_P * KSTEPS * 32 * 16;
  if (t >= total) return;
  int e    = t & 15;
  int lane = (t >> 4) & 31;
  int rest = t >> 9;                      // (layer*OCT_P + ocT)*KSTEPS + kStep
  int kStep = rest % KSTEPS;  rest /= KSTEPS;
  int ocT   = rest % OCT_P;
  int layer = rest / OCT_P;
  int oc = ocT * 16 + (lane & 15);
  int k  = kStep * 32 + a_elem_k(lane, e);
  float w = 0.f;
  if (oc < INF && k < KCONV) {
    const float* wsrc = (layer == 0) ? w_head : (w_blocks + (size_t)(layer - 1) * INF * KCONV);
    w = wsrc[(size_t)oc * KCONV + k];     // k = ic*9 + ky*3 + kx (OIHW row-major)
  }
  apack[t] = (bf16)w;
}

__global__ void pack_last_w(const float* __restrict__ w_last, bf16* __restrict__ apack) {
  int t = blockIdx.x * blockDim.x + threadIdx.x;
  int total = OCT_L * KSTEPS_L * 32 * 16;
  if (t >= total) return;
  int e    = t & 15;
  int lane = (t >> 4) & 31;
  int rest = t >> 9;
  int kStep = rest % KSTEPS_L;
  int ocT   = rest / KSTEPS_L;
  int oc = ocT * 16 + (lane & 15);
  int k  = kStep * 32 + a_elem_k(lane, e);
  float w = 0.f;
  if (oc < OC36 && k < KLAST) w = w_last[(size_t)oc * KLAST + k];
  apack[t] = (bf16)w;
}

// ---------------- bilinear warp + concat -> padded 100ch bf16 input ----------------

__global__ void warp_concat_kernel(const float* __restrict__ f0,
                                   const float* __restrict__ f1,
                                   const float* __restrict__ fd,
                                   const float* __restrict__ flow,
                                   bf16* __restrict__ xin) {
  int t = blockIdx.x * blockDim.x + threadIdx.x;
  if (t >= INF * HWp) return;
  int c = t / HWp;
  int p = t - c * HWp;
  int y = p / WW, x = p - (p / WW) * WW;
  float v;
  if (c < 64) {
    const float* feat = (c < 32) ? f0 : f1;
    int cc = c & 31;
    const float* fx = flow + (size_t)((c < 32) ? 0 : 2) * HWp;
    const float* fy = fx + HWp;
    float px = (float)x + fx[p];
    float py = (float)y + fy[p];
    float x0 = floorf(px), y0 = floorf(py);
    float wx = px - x0,    wy = py - y0;
    int x0i = iclamp((int)x0,     0, WW - 1);
    int x1i = iclamp((int)x0 + 1, 0, WW - 1);
    int y0i = iclamp((int)y0,     0, HH - 1);
    int y1i = iclamp((int)y0 + 1, 0, HH - 1);
    const float* img = feat + (size_t)cc * HWp;
    float I00 = img[y0i * WW + x0i];
    float I01 = img[y0i * WW + x1i];
    float I10 = img[y1i * WW + x0i];
    float I11 = img[y1i * WW + x1i];
    v = I00 * (1.f - wx) * (1.f - wy) + I01 * wx * (1.f - wy)
      + I10 * (1.f - wx) * wy         + I11 * wx * wy;
  } else if (c < 96) {
    v = fd[(size_t)(c - 64) * HWp + p];
  } else {
    v = flow[(size_t)(c - 96) * HWp + p];
  }
  xin[(size_t)c * CHS + (size_t)(y + 1) * PAD + (x + 1)] = (bf16)v;
}

// ---------------- implicit-GEMM 3x3 conv: TDM -> LDS -> bf16 WMMA ----------------
// Block: 224 threads (7 waves) = one output row, one oc-group (64 oc).
// Wave: 2 pixel tiles x 4 oc tiles = 8 WMMA accumulators.
// TDM stages the [100ch][3row][226col] input slab into LDS in one instruction.

__global__ __launch_bounds__(224)
void conv_wmma(const bf16* __restrict__ xin, const bf16* __restrict__ apack,
               const float* __restrict__ alpha, const float* __restrict__ mask,
               bf16* __restrict__ out, int head_mode) {
  extern __shared__ bf16 lds[];             // [100][3][226] + 256 guard zeros
  const int tid  = threadIdx.x;
  const int lane = tid & 31;
  const int wv   = tid >> 5;                // 0..6
  const int row  = blockIdx.x >> 1;         // 0..223
  const int ocg  = blockIdx.x & 1;          // oc 0..63 / 64..127
  const int n    = lane & 15;
  const int kg   = lane >> 4;

  // zero the K>=900 guard slot
  for (int g = tid; g < 256; g += 224) lds[ZOFF + g] = (bf16)0.f;

  if (tid == 0) {
    // --- Tensor DMA descriptor (ISA cdna5 08_async_tensor §8) ---
    uint64_t gaddr = (uint64_t)(uintptr_t)xin + (uint64_t)row * (PAD * 2); // tile @ (ic0,row,x0)
    u32x4 g0;
    g0[0] = 1u;                                        // count=1, no gather
    g0[1] = 0u;                                        // lds_addr = 0 (dynamic LDS base)
    g0[2] = (unsigned)(gaddr & 0xffffffffu);
    g0[3] = (unsigned)((gaddr >> 32) & 0x1ffffffu) | (2u << 30);  // type=2 "image"
    i32x8 g1;
    g1[0] = (1 << 16);                                 // data_size=1 -> 2 bytes
    g1[1] = (int)((unsigned)PAD << 16);                // tensor_dim0[15:0]=226
    g1[2] = (int)((unsigned)PAD << 16);                // dim0 hi=0 | tensor_dim1[15:0]=226
    g1[3] = (int)((unsigned)PAD << 16);                // dim1 hi=0 | tile_dim0=226
    g1[4] = (int)(3u | (100u << 16));                  // tile_dim1=3 | tile_dim2=100
    g1[5] = PAD;                                       // tensor_dim0_stride = 226
    g1[6] = (int)(((unsigned)CHS & 0xffffu) << 16);    // stride0 hi=0 | stride1 lo16 (51076)
    g1[7] = (int)((unsigned)CHS >> 16);                // stride1[47:16]
    i32x4 g2;
    g2[0] = INF;                                       // tensor_dim2 = 100
    g2[1] = 0;                                         // tensor_dim3 unused
    g2[2] = CHS;                                       // tensor_dim2_stride = 51076
    g2[3] = 0;                                         // stride2 hi | tile_dim3=0
    i32x4 g3 = {0, 0, 0, 0};
#if defined(__clang_major__) && __clang_major__ >= 23
    i32x8 gx = {0, 0, 0, 0, 0, 0, 0, 0};
    __builtin_amdgcn_tensor_load_to_lds(g0, g1, g2, g3, gx, 0);
#else
    __builtin_amdgcn_tensor_load_to_lds(g0, g1, g2, g3, 0);
#endif
    __builtin_amdgcn_s_wait_tensorcnt(0);
  }
  __syncthreads();

  const int x0 = wv * 32 + n;               // tile0 pixel x; tile1 = x0+16
  v8f accA[4] = {{0}, {0}, {0}, {0}};
  v8f accB[4] = {{0}, {0}, {0}, {0}};

  // main K loop: k < 896 always valid (no guard select needed)
  for (int ks = 0; ks < KSTEPS - 1; ++ks) {
    v16bf a[4];
#pragma unroll
    for (int j = 0; j < 4; ++j)
      a[j] = *(const v16bf*)(apack +
              ((((size_t)(ocg * 4 + j)) * KSTEPS + ks) * 32 + lane) * 16);
    int kbase = ks * 32 + kg * 8;
    v16bf b0, b1;
#pragma unroll
    for (int e = 0; e < 16; ++e) {
      int off = conv_off(kbase + e + (e & 8));        // (k/3)*223 + k
      b0[e] = lds[off + x0];
      b1[e] = lds[off + x0 + 16];
    }
#pragma unroll
    for (int j = 0; j < 4; ++j) {
      accA[j] = __builtin_amdgcn_wmma_f32_16x16x32_bf16(false, a[j], false, b0,
                                                        (short)0, accA[j], false, false);
      accB[j] = __builtin_amdgcn_wmma_f32_16x16x32_bf16(false, a[j], false, b1,
                                                        (short)0, accB[j], false, false);
    }
  }
  {                                          // peeled final K step: k in [896,928)
    constexpr int ks = KSTEPS - 1;
    v16bf a[4];
#pragma unroll
    for (int j = 0; j < 4; ++j)
      a[j] = *(const v16bf*)(apack +
              ((((size_t)(ocg * 4 + j)) * KSTEPS + ks) * 32 + lane) * 16);
    int kbase = ks * 32 + kg * 8;
    v16bf b0, b1;
#pragma unroll
    for (int e = 0; e < 16; ++e) {
      int k = kbase + e + (e & 8);
      int off = (k < KCONV) ? conv_off(k) : ZOFF;   // K padding -> zero slot
      b0[e] = lds[off + x0];
      b1[e] = lds[off + x0 + 16];
    }
#pragma unroll
    for (int j = 0; j < 4; ++j) {
      accA[j] = __builtin_amdgcn_wmma_f32_16x16x32_bf16(false, a[j], false, b0,
                                                        (short)0, accA[j], false, false);
      accB[j] = __builtin_amdgcn_wmma_f32_16x16x32_bf16(false, a[j], false, b1,
                                                        (short)0, accB[j], false, false);
    }
  }

  float m0 = mask[row * WW + x0];
  float m1 = mask[row * WW + x0 + 16];
#pragma unroll
  for (int j = 0; j < 4; ++j) {
#pragma unroll
    for (int rr = 0; rr < 8; ++rr) {
      int oc = ocg * 64 + j * 16 + kg * 8 + rr;
      if (oc < INF) {
        float v0 = accA[j][rr], v1 = accB[j][rr];
        float al = alpha[oc];
        if (head_mode) {                    // prelu(conv) * mask
          v0 = (v0 >= 0.f) ? v0 : al * v0;  v0 *= m0;
          v1 = (v1 >= 0.f) ? v1 : al * v1;  v1 *= m1;
        } else {                            // prelu(conv * mask)
          v0 *= m0;  v0 = (v0 >= 0.f) ? v0 : al * v0;
          v1 *= m1;  v1 = (v1 >= 0.f) ? v1 : al * v1;
        }
        size_t base = ((size_t)oc * PAD + row + 1) * PAD + 1;
        out[base + x0]      = (bf16)v0;
        out[base + x0 + 16] = (bf16)v1;
      }
    }
  }
}

// ---------------- 500 -> 36 channel mix (w_last einsum) via WMMA ----------------

__global__ __launch_bounds__(256)
void last_wmma(const bf16* __restrict__ layers, const bf16* __restrict__ apack,
               const float* __restrict__ b_last, float* __restrict__ x36) {
  int wave = (blockIdx.x * blockDim.x + threadIdx.x) >> 5;
  int lane = threadIdx.x & 31;
  int ocT = wave / PIXT;
  int pT  = wave - ocT * PIXT;
  int n  = lane & 15;
  int kg = lane >> 4;
  int p  = pT * 16 + n;
  int y  = p / WW;
  int x  = p - y * WW;
  size_t pix = (size_t)(y + 1) * PAD + (x + 1);

  v8f acc = {0};
  for (int ks = 0; ks < KSTEPS_L; ++ks) {
    v16bf a = *(const v16bf*)(apack + (((size_t)ocT * KSTEPS_L + ks) * 32 + lane) * 16);
    v16bf b;
    int kbase = ks * 32 + kg * 8;
#pragma unroll
    for (int e = 0; e < 16; ++e) {
      int k = kbase + e + (e & 8);                    // flat feats channel: layer*100 + c
      b[e] = (k < KLAST) ? layers[(size_t)k * CHS + pix] : (bf16)0.f;
    }
    acc = __builtin_amdgcn_wmma_f32_16x16x32_bf16(false, a, false, b,
                                                  (short)0, acc, false, false);
  }
#pragma unroll
  for (int r = 0; r < 8; ++r) {
    int oc = ocT * 16 + kg * 8 + r;
    if (oc < OC36) x36[(size_t)oc * HWp + p] = acc[r] + b_last[oc];
  }
}

// ---------------- tiny 32 -> 2 mask conv (VALU; 0.06 GFLOP) ----------------

__global__ void mask_conv(const float* __restrict__ x36, const float* __restrict__ w_mask,
                          const float* __restrict__ b_mask, float* __restrict__ smn) {
  int p = blockIdx.x * blockDim.x + threadIdx.x;
  if (p >= HWp) return;
  int y = p / WW, x = p - (p / WW) * WW;
  for (int oc = 0; oc < 2; ++oc) {
    float acc = b_mask[oc];
    for (int ic = 0; ic < NF; ++ic) {
      const float* img = x36 + (size_t)(4 + ic) * HWp;
      const float* wv  = w_mask + (size_t)(oc * NF + ic) * 9;
#pragma unroll
      for (int ky = 0; ky < 3; ++ky)
#pragma unroll
        for (int kx = 0; kx < 3; ++kx) {
          int yy = y + ky - 1, xx = x + kx - 1;
          if ((unsigned)yy < (unsigned)HH && (unsigned)xx < (unsigned)WW)
            acc += img[yy * WW + xx] * wv[ky * 3 + kx];
        }
    }
    smn[(size_t)oc * HWp + p] = acc;
  }
}

// ---------------- 2x half-pixel bilinear upsample + mask output ----------------

__global__ void upsample_out(const float* __restrict__ x36, const float* __restrict__ smn,
                             const float* __restrict__ space_mask, float* __restrict__ out) {
  int t = blockIdx.x * blockDim.x + threadIdx.x;
  if (t >= S2) return;
  int oy = t / W2, ox = t - (t / W2) * W2;
  float sy = (oy + 0.5f) * 0.5f - 0.5f;
  float sx = (ox + 0.5f) * 0.5f - 0.5f;
  float fy = floorf(sy), fx = floorf(sx);
  float wy = sy - fy,    wx = sx - fx;
  int y0 = iclamp((int)fy,     0, HH - 1);
  int y1 = iclamp((int)fy + 1, 0, HH - 1);
  int x0 = iclamp((int)fx,     0, WW - 1);
  int x1 = iclamp((int)fx + 1, 0, WW - 1);
  float w00 = (1.f - wx) * (1.f - wy), w01 = wx * (1.f - wy);
  float w10 = (1.f - wx) * wy,         w11 = wx * wy;
  int i00 = y0 * WW + x0, i01 = y0 * WW + x1;
  int i10 = y1 * WW + x0, i11 = y1 * WW + x1;
  for (int c = 0; c < OC36; ++c) {
    const float* img = x36 + (size_t)c * HWp;
    out[(size_t)c * S2 + t] =
        img[i00] * w00 + img[i01] * w01 + img[i10] * w10 + img[i11] * w11;
  }
  float s0 = smn[i00] * w00 + smn[i01] * w01 + smn[i10] * w10 + smn[i11] * w11;
  const float* sm1 = smn + HWp;
  float s1 = sm1[i00] * w00 + sm1[i01] * w01 + sm1[i10] * w10 + sm1[i11] * w11;
  float mu = space_mask[(oy >> 1) * WW + (ox >> 1)];
  out[(size_t)OC36 * S2 + t] = (s0 > s1 ? 1.f : 0.f) * mu;
}

// ---------------- launch ----------------

extern "C" void kernel_launch(void* const* d_in, const int* in_sizes, int n_in,
                              void* d_out, int out_size, void* d_ws, size_t ws_size,
                              hipStream_t stream) {
  (void)in_sizes; (void)n_in; (void)out_size; (void)ws_size;
  const float* feat_0     = (const float*)d_in[0];
  const float* feat_1     = (const float*)d_in[1];
  const float* feat_dense = (const float*)d_in[2];
  const float* flow       = (const float*)d_in[3];
  const float* space_mask = (const float*)d_in[4];
  const float* w_head     = (const float*)d_in[5];
  const float* a_head     = (const float*)d_in[6];
  const float* w_blocks   = (const float*)d_in[7];
  const float* a_blocks   = (const float*)d_in[8];
  const float* w_last     = (const float*)d_in[9];
  const float* b_last     = (const float*)d_in[10];
  const float* w_mask     = (const float*)d_in[11];
  const float* b_mask     = (const float*)d_in[12];
  float* out = (float*)d_out;

  char* ws = (char*)d_ws;
  size_t off = 0;
  auto alloc = [&](size_t bytes) -> void* {
    void* p = ws + off;
    off = (off + bytes + 255) & ~(size_t)255;
    return p;
  };
  const size_t layerStride = (size_t)INF * CHS;            // padded [100][226][226]
  const size_t xbfBytes    = 6 * layerStride * sizeof(bf16);
  bf16*  xbf       = (bf16*)alloc(xbfBytes);               // xin + 5 layer outs
  bf16*  apackConv = (bf16*)alloc((size_t)5 * OCT_P * KSTEPS * 32 * 16 * sizeof(bf16));
  bf16*  apackLast = (bf16*)alloc((size_t)OCT_L * KSTEPS_L * 32 * 16 * sizeof(bf16));
  float* x36       = (float*)alloc((size_t)OC36 * HWp * sizeof(float));
  float* smn       = (float*)alloc((size_t)2 * HWp * sizeof(float));

  // zero guard rows/cols of all padded activation buffers (interiors rewritten below)
  hipMemsetAsync(xbf, 0, xbfBytes, stream);

  { int nT = 5 * OCT_P * KSTEPS * 32 * 16;
    pack_conv_w<<<(nT + 255) / 256, 256, 0, stream>>>(w_head, w_blocks, apackConv); }
  { int nT = OCT_L * KSTEPS_L * 32 * 16;
    pack_last_w<<<(nT + 255) / 256, 256, 0, stream>>>(w_last, apackLast); }
  { int nT = INF * HWp;
    warp_concat_kernel<<<(nT + 255) / 256, 256, 0, stream>>>(feat_0, feat_1, feat_dense,
                                                             flow, xbf); }
  for (int l = 0; l < 5; ++l) {
    const bf16* in  = xbf + (size_t)l * layerStride;
    bf16*       ob  = xbf + (size_t)(l + 1) * layerStride;
    const bf16* ap  = apackConv + (size_t)l * OCT_P * KSTEPS * 32 * 16;
    const float* al = (l == 0) ? a_head : (a_blocks + (size_t)(l - 1) * INF);
    conv_wmma<<<HH * 2, 224, LDS_BYTES, stream>>>(in, ap, al, space_mask, ob,
                                                  l == 0 ? 1 : 0);
  }
  const int lastBlocks = (OCT_L * PIXT) / 8;
  last_wmma<<<lastBlocks, 256, 0, stream>>>(xbf + layerStride, apackLast, b_last, x36);
  mask_conv<<<(HWp + 255) / 256, 256, 0, stream>>>(x36, w_mask, b_mask, smn);
  upsample_out<<<(S2 + 255) / 256, 256, 0, stream>>>(x36, smn, space_mask, out);
}
